// CrossAttentionBlock_9423158248059
// MI455X (gfx1250) — compile-verified, matrix operand order checked by reading, not run on hardware
//
#include <hip/hip_runtime.h>
#include <hip/hip_bf16.h>

// ---------------------------------------------------------------------------
// CrossAttentionBlock for MI455X (gfx1250).
// B=32, T=1024, C=512, COND_CH=256, L=252, S=128, BT=32768.
// GEMMs: v_wmma_f32_16x16x32_f16, double-buffered LDS fed by
// global_load_async_to_lds_b128 (ASYNCcnt pipeline). Attention collapsed to
// rank-2 form (cond is conv_w[s]*pad + conv_b[s] -> k/v are rank-2 in s).
// ---------------------------------------------------------------------------

#define Bb    32
#define Tt    1024
#define Cc    512
#define BT    32768          // Bb*Tt
#define Ll    252
#define CONDC 256
#define Ss    128
#define LN_EPS 1e-5f

typedef __attribute__((ext_vector_type(16))) _Float16 v16h;
typedef __attribute__((ext_vector_type(8)))  _Float16 v8h;
typedef __attribute__((ext_vector_type(4)))  _Float16 v4h;
typedef __attribute__((ext_vector_type(8)))  float    v8f;

// ---------------------------------------------------------------- utilities

__device__ __forceinline__ float wave_sum(float v) {
#pragma unroll
  for (int m = 16; m >= 1; m >>= 1) v += __shfl_xor(v, m, 32);
  return v;
}
__device__ __forceinline__ float wave_max(float v) {
#pragma unroll
  for (int m = 16; m >= 1; m >>= 1) v = fmaxf(v, __shfl_xor(v, m, 32));
  return v;
}

// CDNA5 async copy: memory -> LDS, tracked by ASYNCcnt (no VGPR staging).
__device__ __forceinline__ void async_copy_b128(unsigned lds_off,
                                                const _Float16* gptr) {
  asm volatile("global_load_async_to_lds_b128 %0, %1, off"
               :
               : "v"(lds_off), "v"(gptr)
               : "memory");
}
__device__ __forceinline__ void wait_async_le4() {
  asm volatile("s_wait_asynccnt 0x4" ::: "memory");
}
__device__ __forceinline__ void wait_async_0() {
  asm volatile("s_wait_asynccnt 0x0" ::: "memory");
}

// ------------------------------------------------------------ prep (rank-2)
// pk[b,c] = sum_j pad[b,j]*Wk[c,j], pv likewise; rwk[c]=sum_j Wk[c,j], rwv.
__global__ __launch_bounds__(256) void prep_kernel(
    const float* __restrict__ cond_emb, const float* __restrict__ Wk,
    const float* __restrict__ Wv, float* __restrict__ pk,
    float* __restrict__ pv, float* __restrict__ rwk, float* __restrict__ rwv) {
  const int idx = blockIdx.x * 256 + threadIdx.x;       // 0 .. 16383
  const int b = idx >> 9;
  const int c = idx & 511;
  float s1 = 0.f, s2 = 0.f;
  const float* wkr = Wk + c * CONDC;
  const float* wvr = Wv + c * CONDC;
  const float* emb = cond_emb + b * Ll;
#pragma unroll 4
  for (int j = 0; j < CONDC; ++j) {
    const float p = emb[(j + (Ll - 2)) % Ll];           // circular pad by 2
    s1 += p * wkr[j];
    s2 += p * wvr[j];
  }
  pk[b * Cc + c] = s1;
  pv[b * Cc + c] = s2;
  if (b == 0) {
    float r1 = 0.f, r2 = 0.f;
#pragma unroll 4
    for (int j = 0; j < CONDC; ++j) { r1 += wkr[j]; r2 += wvr[j]; }
    rwk[c] = r1;
    rwv[c] = r2;
  }
}

// -------------------------------------------------------------- f32 -> f16
__global__ __launch_bounds__(256) void cvt_f16_kernel(
    const float4* __restrict__ in, v4h* __restrict__ out) {
  const int i = blockIdx.x * 256 + threadIdx.x;
  const float4 f = in[i];
  v4h o;
  o.x = (_Float16)f.x; o.y = (_Float16)f.y;
  o.z = (_Float16)f.z; o.w = (_Float16)f.w;
  out[i] = o;
}

// ------------------------------------------------------------- WMMA GEMM
// C[M,N] = A[M,K] @ Bw[N,K]^T (+bias, +epilogue). f16 in, f32 accumulate.
// Block: 256 threads = 8 waves, block tile 128x128; wave tile 64x32
// (4x2 WMMA accumulators -> 8 WMMAs per 12 ds_load_b128).
// LDS double-buffered, filled with global_load_async_to_lds_b128.
// EPI 0: f32 store (bias).  EPI 1: exact GELU then f16 store (bias).
#define LDSP 40   // LDS row pitch in halves (bank-conflict padding)

template <int EPI>
__global__ __launch_bounds__(256) void gemm_wmma_kernel(
    const _Float16* __restrict__ A, const _Float16* __restrict__ Bw,
    const float* __restrict__ bias, void* __restrict__ outp,
    const int K, const int N) {
  __shared__ __align__(16) _Float16 Al[2][128 * LDSP];
  __shared__ __align__(16) _Float16 Bl[2][128 * LDSP];

  const int tid = threadIdx.x;
  const int m0 = blockIdx.x * 128;
  const int n0 = blockIdx.y * 128;

  const int w  = tid >> 5;
  const int l  = tid & 31;
  const int wm = (w >> 2) * 64;   // wave row origin in block tile (2 rows)
  const int wn = (w & 3) * 32;    // wave col origin in block tile (4 cols)
  const int r  = l & 15;          // lane row/col within 16
  const int h  = l >> 4;          // lane half

  v8f acc[4][2];
#pragma unroll
  for (int mi = 0; mi < 4; ++mi)
#pragma unroll
    for (int ni = 0; ni < 2; ++ni)
#pragma unroll
      for (int e = 0; e < 8; ++e) acc[mi][ni][e] = 0.0f;

  // Tile staging: 128 rows x 32 halves each for A and B; 2 threads per row,
  // 16 halves (2 x b128) per thread.
  const int srow = tid >> 1;
  const int scol = (tid & 1) * 16;
  const _Float16* ag0 = A + (size_t)(m0 + srow) * K + scol;
  const _Float16* bg0 = Bw + (size_t)(n0 + srow) * K + scol;

  auto stage = [&](int k0, int buf) {
    const unsigned la = (unsigned)(size_t)&Al[buf][srow * LDSP + scol];
    const unsigned lb = (unsigned)(size_t)&Bl[buf][srow * LDSP + scol];
    async_copy_b128(la,      ag0 + k0);
    async_copy_b128(la + 16, ag0 + k0 + 8);
    async_copy_b128(lb,      bg0 + k0);
    async_copy_b128(lb + 16, bg0 + k0 + 8);
  };

  stage(0, 0);
  int cur = 0;
  for (int k0 = 0; k0 < K; k0 += 32) {
    // Readers of buf[cur^1] (previous iteration) must be done before its
    // async overwrite below.
    __syncthreads();
    const bool has_next = (k0 + 32) < K;
    if (has_next) {
      stage(k0 + 32, cur ^ 1);
      wait_async_le4();   // 4 in flight (next tile); buf[cur] writes complete
    } else {
      wait_async_0();
    }
    __syncthreads();      // all waves' copies into buf[cur] are visible

    // ISA 16-bit A 16x32 layout: lane(l) row=r; halves 0..7 = K[8h..8h+7],
    // halves 8..15 = K[16+8h..16+8h+7]. B[N,K] rows give B^T columns with
    // mirrored striping -> identical LDS addressing.
    v16h af[4], bf[2];
#pragma unroll
    for (int mi = 0; mi < 4; ++mi) {
      const _Float16* p = &Al[cur][(wm + mi * 16 + r) * LDSP + h * 8];
      v8h lo = *(const v8h*)p;
      v8h hi = *(const v8h*)(p + 16);
      af[mi] = __builtin_shufflevector(lo, hi, 0, 1, 2, 3, 4, 5, 6, 7, 8, 9,
                                       10, 11, 12, 13, 14, 15);
    }
#pragma unroll
    for (int ni = 0; ni < 2; ++ni) {
      const _Float16* p = &Bl[cur][(wn + ni * 16 + r) * LDSP + h * 8];
      v8h lo = *(const v8h*)p;
      v8h hi = *(const v8h*)(p + 16);
      bf[ni] = __builtin_shufflevector(lo, hi, 0, 1, 2, 3, 4, 5, 6, 7, 8, 9,
                                       10, 11, 12, 13, 14, 15);
    }
#pragma unroll
    for (int mi = 0; mi < 4; ++mi)
#pragma unroll
      for (int ni = 0; ni < 2; ++ni)
        acc[mi][ni] = __builtin_amdgcn_wmma_f32_16x16x32_f16(
            false, af[mi], false, bf[ni], (short)0, acc[mi][ni], false, false);
    cur ^= 1;
  }

  // C/D layout: lane l -> col r, rows h*8+vr.
#pragma unroll
  for (int mi = 0; mi < 4; ++mi)
#pragma unroll
    for (int ni = 0; ni < 2; ++ni) {
      const int gcol = n0 + wn + ni * 16 + r;
      const float bz = bias[gcol];
#pragma unroll
      for (int vr = 0; vr < 8; ++vr) {
        const int grow = m0 + wm + mi * 16 + h * 8 + vr;
        const float vx = acc[mi][ni][vr] + bz;
        if (EPI == 0) {
          ((float*)outp)[(size_t)grow * N + gcol] = vx;
        } else {
          const float g = 0.5f * vx * (1.0f + erff(vx * 0.70710678118654752f));
          ((_Float16*)outp)[(size_t)grow * N + gcol] = (_Float16)g;
        }
      }
    }
}

// ---------------------------------------------- attention (rank-2) + LN1
// One wave32 per query row. scores[s] = scale*(convw[s]*d1+convb[s]*d2+d3),
// attn_out = a1*pv[b] + a2*rwv + bv;  query = LN1(q + attn_out).
__global__ __launch_bounds__(256) void attn_ln1_kernel(
    const float* __restrict__ q, const float* __restrict__ pk,
    const float* __restrict__ pv, const float* __restrict__ rwk,
    const float* __restrict__ rwv, const float* __restrict__ bk,
    const float* __restrict__ bv, const float* __restrict__ convw,
    const float* __restrict__ convb, const float* __restrict__ lnw,
    const float* __restrict__ lnb, float* __restrict__ query,
    _Float16* __restrict__ query_h) {
  const int gw   = blockIdx.x * 8 + (threadIdx.x >> 5);
  const int lane = threadIdx.x & 31;
  const int b    = gw >> 10;                    // row / T
  const size_t rowoff = (size_t)gw * Cc;
  const int c0 = lane * 16;

  float qv[16];
  float d1 = 0.f, d2 = 0.f, d3 = 0.f;
#pragma unroll
  for (int i = 0; i < 16; ++i) {
    const int c = c0 + i;
    const float qq = q[rowoff + c];
    qv[i] = qq;
    d1 += qq * pk[b * Cc + c];
    d2 += qq * rwk[c];
    d3 += qq * bk[c];
  }
  d1 = wave_sum(d1); d2 = wave_sum(d2); d3 = wave_sum(d3);

  const float scale = 0.044194173824159216f;    // 1/sqrt(512)
  float sc[4];
#pragma unroll
  for (int i = 0; i < 4; ++i) {
    const int s = lane + 32 * i;
    sc[i] = scale * (convw[s] * d1 + convb[s] * d2 + d3);
  }
  float mx = fmaxf(fmaxf(sc[0], sc[1]), fmaxf(sc[2], sc[3]));
  mx = wave_max(mx);
  float wsum = 0.f, a1 = 0.f, a2 = 0.f;
#pragma unroll
  for (int i = 0; i < 4; ++i) {
    const int s = lane + 32 * i;
    const float e = __expf(sc[i] - mx);
    wsum += e;
    a1 += e * convw[s];
    a2 += e * convb[s];
  }
  wsum = wave_sum(wsum); a1 = wave_sum(a1); a2 = wave_sum(a2);
  const float inv = 1.0f / wsum;
  a1 *= inv; a2 *= inv;

  float y[16];
  float psum = 0.f;
#pragma unroll
  for (int i = 0; i < 16; ++i) {
    const int c = c0 + i;
    const float t = qv[i] + a1 * pv[b * Cc + c] + a2 * rwv[c] + bv[c];
    y[i] = t;
    psum += t;
  }
  const float mu = wave_sum(psum) * (1.0f / (float)Cc);
  float pvar = 0.f;
#pragma unroll
  for (int i = 0; i < 16; ++i) {
    const float d = y[i] - mu;
    pvar += d * d;
  }
  const float var = wave_sum(pvar) * (1.0f / (float)Cc);
  const float rs = rsqrtf(var + LN_EPS);
#pragma unroll
  for (int i = 0; i < 16; ++i) {
    const int c = c0 + i;
    const float o = (y[i] - mu) * rs * lnw[c] + lnb[c];
    query[rowoff + c]   = o;
    query_h[rowoff + c] = (_Float16)o;
  }
}

// ------------------------------------------------- final LN2 + residual
__global__ __launch_bounds__(256) void final_ln2_kernel(
    const float* __restrict__ query, const float* __restrict__ ff,
    const float* __restrict__ x, const float* __restrict__ lnw,
    const float* __restrict__ lnb, float* __restrict__ out) {
  const int gw   = blockIdx.x * 8 + (threadIdx.x >> 5);
  const int lane = threadIdx.x & 31;
  const size_t rowoff = (size_t)gw * Cc;
  const int c0 = lane * 16;

  float y[16];
  float psum = 0.f;
#pragma unroll
  for (int i = 0; i < 16; ++i) {
    const int c = c0 + i;
    const float t = query[rowoff + c] + ff[rowoff + c];
    y[i] = t;
    psum += t;
  }
  const float mu = wave_sum(psum) * (1.0f / (float)Cc);
  float pvar = 0.f;
#pragma unroll
  for (int i = 0; i < 16; ++i) {
    const float d = y[i] - mu;
    pvar += d * d;
  }
  const float var = wave_sum(pvar) * (1.0f / (float)Cc);
  const float rs = rsqrtf(var + LN_EPS);
#pragma unroll
  for (int i = 0; i < 16; ++i) {
    const int c = c0 + i;
    out[rowoff + c] =
        (y[i] - mu) * rs * lnw[c] + lnb[c] + x[rowoff + c];
  }
}

// --------------------------------------------------------------- launcher
extern "C" void kernel_launch(void* const* d_in, const int* in_sizes, int n_in,
                              void* d_out, int out_size, void* d_ws,
                              size_t ws_size, hipStream_t stream) {
  const float* x        = (const float*)d_in[0];
  const float* cond_emb = (const float*)d_in[1];
  const float* Wq       = (const float*)d_in[2];
  const float* bq       = (const float*)d_in[3];
  const float* Wk       = (const float*)d_in[4];
  const float* bk       = (const float*)d_in[5];
  const float* Wv       = (const float*)d_in[6];
  const float* bv       = (const float*)d_in[7];
  const float* convw    = (const float*)d_in[8];
  const float* convb    = (const float*)d_in[9];
  const float* ln1w     = (const float*)d_in[10];
  const float* ln1b     = (const float*)d_in[11];
  const float* W1       = (const float*)d_in[12];
  const float* b1       = (const float*)d_in[13];
  const float* W2       = (const float*)d_in[14];
  const float* b2       = (const float*)d_in[15];
  const float* ln2w     = (const float*)d_in[16];
  const float* ln2b     = (const float*)d_in[17];
  float* out = (float*)d_out;

  // ---- workspace carve (aliasing: ff reuses q, query_h reuses x_h) ----
  char* ws = (char*)d_ws;
  auto carve = [&](size_t bytes) {
    char* p = ws;
    ws += (bytes + 255) & ~(size_t)255;
    return p;
  };
  float*    qbuf   = (float*)carve((size_t)BT * Cc * 4);       // q, later ff
  float*    query  = (float*)carve((size_t)BT * Cc * 4);
  _Float16* h_h    = (_Float16*)carve((size_t)BT * 2 * Cc * 2);
  _Float16* x_h    = (_Float16*)carve((size_t)BT * Cc * 2);    // later query_h
  _Float16* Wq_h   = (_Float16*)carve((size_t)Cc * Cc * 2);
  _Float16* W1_h   = (_Float16*)carve((size_t)2 * Cc * Cc * 2);
  _Float16* W2_h   = (_Float16*)carve((size_t)Cc * 2 * Cc * 2);
  float*    pk     = (float*)carve((size_t)Bb * Cc * 4);
  float*    pv     = (float*)carve((size_t)Bb * Cc * 4);
  float*    rwk    = (float*)carve((size_t)Cc * 4);
  float*    rwv    = (float*)carve((size_t)Cc * 4);
  float*    ffbuf  = qbuf;                  // alias: q dead after attn
  _Float16* query_h = x_h;                  // alias: x_h dead after q-GEMM

  // 1) rank-2 conditioner precompute
  prep_kernel<<<64, 256, 0, stream>>>(cond_emb, Wk, Wv, pk, pv, rwk, rwv);

  // 2) f32 -> f16 operand conversions
  cvt_f16_kernel<<<(BT * Cc) / 1024, 256, 0, stream>>>(
      (const float4*)x, (v4h*)x_h);
  cvt_f16_kernel<<<(Cc * Cc) / 1024, 256, 0, stream>>>(
      (const float4*)Wq, (v4h*)Wq_h);
  cvt_f16_kernel<<<(2 * Cc * Cc) / 1024, 256, 0, stream>>>(
      (const float4*)W1, (v4h*)W1_h);
  cvt_f16_kernel<<<(2 * Cc * Cc) / 1024, 256, 0, stream>>>(
      (const float4*)W2, (v4h*)W2_h);

  // 3) q = x @ Wq^T + bq          [32768,512] x [512,512]
  gemm_wmma_kernel<0><<<dim3(BT / 128, Cc / 128), 256, 0, stream>>>(
      x_h, Wq_h, bq, (void*)qbuf, Cc, Cc);

  // 4) attention (rank-2 collapsed) + LN1
  attn_ln1_kernel<<<BT / 8, 256, 0, stream>>>(
      qbuf, pk, pv, rwk, rwv, bk, bv, convw, convb, ln1w, ln1b, query,
      query_h);

  // 5) h = GELU(query @ W1^T + b1)   [32768,512] x [1024,512] -> f16
  gemm_wmma_kernel<1><<<dim3(BT / 128, (2 * Cc) / 128), 256, 0, stream>>>(
      query_h, W1_h, b1, (void*)h_h, Cc, 2 * Cc);

  // 6) ff = h @ W2^T + b2            [32768,1024] x [512,1024]
  gemm_wmma_kernel<0><<<dim3(BT / 128, Cc / 128), 256, 0, stream>>>(
      h_h, W2_h, b2, (void*)ffbuf, 2 * Cc, Cc);

  // 7) out = LN2(query + ff) + x
  final_ln2_kernel<<<BT / 8, 256, 0, stream>>>(query, ffbuf, x, ln2w, ln2b,
                                               out);
}
